// CausalSelfAttention_80994493268532
// MI455X (gfx1250) — matrix operand order repaired and checked by reading, not compiled, not run on hardware
//
#include <hip/hip_runtime.h>
#include <hip/hip_bf16.h>

#define NB 2
#define NT 2048
#define NC 1024
#define NH 16
#define HD 64
#define NP 1024
#define NS (NP + NT)   // 3072
#define NM (NB * NT)   // 4096

typedef _Float16 v16h __attribute__((ext_vector_type(16)));
typedef _Float16 v8h  __attribute__((ext_vector_type(8)));
typedef float    v8f  __attribute__((ext_vector_type(8)));

// ---- WMMA fragment loaders (CDNA5 16x16x32 f16 layouts, CDNA5 ISA) ----

// A-matrix 16x32 f16: lanes 0-15 rows 0-15 with K 0..7 & 16..23; lanes 16-31 K 8..15 & 24..31
__device__ __forceinline__ v16h load_a_frag(const _Float16* A, int lda,
                                            int mBase, int kBase, int lane) {
  int row  = lane & 15;
  int koff = (lane < 16) ? 0 : 8;
  const _Float16* p = A + (size_t)(mBase + row) * lda + (kBase + koff);
  v8h lo = *(const v8h*)(p);
  v8h hi = *(const v8h*)(p + 16);
  v16h a;
#pragma unroll
  for (int i = 0; i < 8; ++i) { a[i] = lo[i]; a[i + 8] = hi[i]; }
  return a;
}

// B-matrix 32x16 f16 from "BT" storage BT[n][k] (row stride ldb):
// lanes 0-15: col n, K 0..7 & 16..23; lanes 16-31: K 8..15 & 24..31
__device__ __forceinline__ v16h load_b_frag(const _Float16* BT, int ldb,
                                            int nBase, int kBase, int lane) {
  int col = lane & 15;
  int kh  = (lane < 16) ? 0 : 8;
  const _Float16* p = BT + (size_t)(nBase + col) * ldb + (kBase + kh);
  v8h lo = *(const v8h*)(p);
  v8h hi = *(const v8h*)(p + 16);
  v16h b;
#pragma unroll
  for (int i = 0; i < 8; ++i) { b[i] = lo[i]; b[i + 8] = hi[i]; }
  return b;
}

// ---- DPP 16-lane butterfly max (no LDS; stays within each 16-lane half) ----
template <int CTRL>
__device__ __forceinline__ float dpp_fmax_step(float x) {
  int perm = __builtin_amdgcn_update_dpp(__float_as_int(x), __float_as_int(x),
                                         CTRL, 0xF, 0xF, true);
  return fmaxf(x, __int_as_float(perm));
}
__device__ __forceinline__ float dpp_max16(float x) {
  x = dpp_fmax_step<0xB1>(x);   // quad_perm [1,0,3,2] : xor 1
  x = dpp_fmax_step<0x4E>(x);   // quad_perm [2,3,0,1] : xor 2
  x = dpp_fmax_step<0x141>(x);  // row_half_mirror     : merge quads within 8
  x = dpp_fmax_step<0x140>(x);  // row_mirror          : merge halves within 16
  return x;
}

// ---- precision conversion / staging kernels ----

__global__ __launch_bounds__(256) void convert_x_kernel(const float* __restrict__ x,
                                                        _Float16* __restrict__ xh) {
  size_t i = (size_t)blockIdx.x * blockDim.x + threadIdx.x;
  if (i < (size_t)NM * NC) xh[i] = (_Float16)x[i];
}

__global__ __launch_bounds__(256) void transpose_w_kernel(const float* __restrict__ W,
                                                          _Float16* __restrict__ WT) {
  size_t i = (size_t)blockIdx.x * blockDim.x + threadIdx.x;
  if (i < (size_t)NC * NC) {
    int n = (int)(i >> 10);
    int k = (int)(i & 1023);
    WT[i] = (_Float16)W[(size_t)k * NC + n];  // WT[n][k] = W[k][n]
  }
}

// copy past K/V into fp32 output caches + f16 scratch (k natural, v transposed)
__global__ __launch_bounds__(256) void copy_past_kernel(
    const float* __restrict__ pk, const float* __restrict__ pv,
    float* __restrict__ kF, float* __restrict__ vF,
    _Float16* __restrict__ kH, _Float16* __restrict__ vTH) {
  size_t i = (size_t)blockIdx.x * blockDim.x + threadIdx.x;
  if (i >= (size_t)NB * NH * NP * HD) return;
  int d  = (int)(i & 63);
  int p  = (int)((i >> 6) & 1023);
  int bh = (int)(i >> 16);  // 64*1024 = 2^16
  size_t dst = ((size_t)bh * NS + p) * HD + d;
  float kv = pk[i], vv = pv[i];
  kF[dst] = kv;
  kH[dst] = (_Float16)kv;
  vF[dst] = vv;
  vTH[((size_t)bh * HD + d) * NS + p] = (_Float16)vv;
}

// ---- WMMA GEMM, 1x4 register blocking: each wave computes a 16x64 strip ----
// Y[m][n] = A[m][:] . BT[n][:] + bias[n]
// MODE 0: Q  -> f16 scratch [B,H,T,hd], pre-scaled by 1/sqrt(hd)
// MODE 1: K  -> fp32 cache rows [P..S) + f16 scratch same layout
// MODE 2: V  -> fp32 cache rows [P..S) + f16 transposed scratch [B,H,hd,S]
// MODE 3: Y  -> fp32 [M,C]
template <int MODE>
__global__ __launch_bounds__(256) void gemm_kernel(
    const _Float16* __restrict__ A, const _Float16* __restrict__ BT,
    const float* __restrict__ bias, float* __restrict__ outF,
    _Float16* __restrict__ outH) {
  int lane  = threadIdx.x & 31;
  int w     = (blockIdx.x * blockDim.x + threadIdx.x) >> 5;
  int ng    = w & 15;   // N = 1024 -> 16 strips of 64
  int tm    = w >> 4;   // M = 4096 -> 256 tiles of 16
  int mBase = tm << 4, nBase = ng << 6;

  v8f c0 = {}, c1 = {}, c2 = {}, c3 = {};
#pragma unroll 2
  for (int k0 = 0; k0 < NC; k0 += 32) {
    v16h a  = load_a_frag(A, NC, mBase, k0, lane);
    v16h b0 = load_b_frag(BT, NC, nBase,      k0, lane);
    v16h b1 = load_b_frag(BT, NC, nBase + 16, k0, lane);
    v16h b2 = load_b_frag(BT, NC, nBase + 32, k0, lane);
    v16h b3 = load_b_frag(BT, NC, nBase + 48, k0, lane);
    c0 = __builtin_amdgcn_wmma_f32_16x16x32_f16(false, a, false, b0, (short)0, c0, false, false);
    c1 = __builtin_amdgcn_wmma_f32_16x16x32_f16(false, a, false, b1, (short)0, c1, false, false);
    c2 = __builtin_amdgcn_wmma_f32_16x16x32_f16(false, a, false, b2, (short)0, c2, false, false);
    c3 = __builtin_amdgcn_wmma_f32_16x16x32_f16(false, a, false, b3, (short)0, c3, false, false);
  }

  int col = lane & 15;
  int rowOff = (lane < 16) ? 0 : 8;
  int h = nBase >> 6;  // all four tiles share the same head
  v8f acc[4] = {c0, c1, c2, c3};
#pragma unroll
  for (int j = 0; j < 4; ++j) {
    int n = nBase + (j << 4) + col;
    float bb = bias[n];
    int d = (j << 4) + col;
#pragma unroll
    for (int i = 0; i < 8; ++i) {
      int m = mBase + rowOff + i;
      float val = acc[j][i] + bb;
      int b_ = m >> 11;     // / NT
      int t  = m & (NT - 1);
      if constexpr (MODE == 0) {
        outH[(((size_t)b_ * NH + h) * NT + t) * HD + d] = (_Float16)(val * 0.125f);
      } else if constexpr (MODE == 1) {
        size_t idx = (((size_t)b_ * NH + h) * NS + NP + t) * HD + d;
        outF[idx] = val;
        outH[idx] = (_Float16)val;
      } else if constexpr (MODE == 2) {
        size_t idxF = (((size_t)b_ * NH + h) * NS + NP + t) * HD + d;
        outF[idxF] = val;
        outH[(((size_t)b_ * NH + h) * HD + d) * NS + (NP + t)] = (_Float16)val;
      } else {
        outF[(size_t)m * NC + n] = val;
      }
    }
  }
}

// ---- flash attention: one wave per (b,h,16-query block), online softmax ----
// row max: DPP butterflies (VALU, co-executes with WMMA)
// row sum: extra WMMA against a ones matrix (reuses the staged P fragment)
__global__ __launch_bounds__(256) void attn_kernel(
    const _Float16* __restrict__ qh, const _Float16* __restrict__ kh,
    const _Float16* __restrict__ vth, _Float16* __restrict__ atth) {
  __shared__ _Float16 plds[8 * 16 * 32];  // one 16x32 P tile per wave
  int lane = threadIdx.x & 31;
  int wib  = threadIdx.x >> 5;
  int w    = blockIdx.x * 8 + wib;
  int bh   = w >> 7;           // 128 query blocks per (b,h)
  int t0   = (w & 127) << 4;

  const _Float16* qb = qh  + (size_t)bh * NT * HD;
  const _Float16* kb = kh  + (size_t)bh * NS * HD;   // BT for Q.K^T: BT[s][d]
  const _Float16* vb = vth + (size_t)bh * HD * NS;   // BT for P.V:   BT[d][s]
  _Float16* pl = &plds[wib * (16 * 32)];

  v16h aq0 = load_a_frag(qb, HD, t0, 0, lane);
  v16h aq1 = load_a_frag(qb, HD, t0, 32, lane);

  v16h onesB;
#pragma unroll
  for (int i = 0; i < 16; ++i) onesB[i] = (_Float16)1.0f;

  int col = lane & 15;
  int rowOff = (lane < 16) ? 0 : 8;

  float mrow[8], lrow[8];
  v8f o0 = {}, o1 = {}, o2 = {}, o3 = {};
#pragma unroll
  for (int i = 0; i < 8; ++i) { mrow[i] = -3.0e38f; lrow[i] = 0.0f; }

  const int sEnd = t0 + 16 + NP;  // last visible key is t0+15+NP
  for (int s0 = 0; s0 < sEnd; s0 += 32) {
    // prefetch next chunk: 32 lanes x 128B lines cover the 4KB K chunk
    __builtin_prefetch(kb + (size_t)(s0 + 32) * HD + lane * 64, 0, 3);
    __builtin_prefetch(vb + (size_t)lane * NS + (s0 + 32), 0, 3);
    __builtin_prefetch(vb + (size_t)(lane + 32) * NS + (s0 + 32), 0, 3);

    // --- scores: two 16x16 tiles covering keys [s0, s0+32) ---
    v8f c0 = {}, c1 = {};
    {
      v16h b00 = load_b_frag(kb, HD, s0, 0, lane);
      v16h b01 = load_b_frag(kb, HD, s0, 32, lane);
      c0 = __builtin_amdgcn_wmma_f32_16x16x32_f16(false, aq0, false, b00, (short)0, c0, false, false);
      c0 = __builtin_amdgcn_wmma_f32_16x16x32_f16(false, aq1, false, b01, (short)0, c0, false, false);
      v16h b10 = load_b_frag(kb, HD, s0 + 16, 0, lane);
      v16h b11 = load_b_frag(kb, HD, s0 + 16, 32, lane);
      c1 = __builtin_amdgcn_wmma_f32_16x16x32_f16(false, aq0, false, b10, (short)0, c1, false, false);
      c1 = __builtin_amdgcn_wmma_f32_16x16x32_f16(false, aq1, false, b11, (short)0, c1, false, false);
    }
    // --- causal mask (branch-free; only boundary chunks need it) ---
    if ((s0 + 31) > (t0 + NP)) {
#pragma unroll
      for (int i = 0; i < 8; ++i) {
        int lim = t0 + rowOff + i + NP;
        c0[i] = ((s0 + col)      <= lim) ? c0[i] : -3.0e38f;
        c1[i] = ((s0 + 16 + col) <= lim) ? c1[i] : -3.0e38f;
      }
    }
    // --- online softmax: DPP row max, exp, stage P tile in LDS ---
    float fscale[8];
#pragma unroll
    for (int i = 0; i < 8; ++i) {
      float v  = dpp_max16(fmaxf(c0[i], c1[i]));
      float nm = fmaxf(mrow[i], v);
      fscale[i] = __expf(mrow[i] - nm);
      mrow[i]   = nm;
      float p0 = __expf(c0[i] - nm);
      float p1 = __expf(c1[i] - nm);
      int r = rowOff + i;
      pl[r * 32 + col]      = (_Float16)p0;
      pl[r * 32 + 16 + col] = (_Float16)p1;
      o0[i] *= fscale[i]; o1[i] *= fscale[i];
      o2[i] *= fscale[i]; o3[i] *= fscale[i];
    }
    // --- P (16x32, via LDS): one WMMA for row sums + four for P.V ---
    v16h ap = load_a_frag(pl, 32, 0, 0, lane);
    v8f cs = {};
    cs = __builtin_amdgcn_wmma_f32_16x16x32_f16(false, ap, false, onesB, (short)0, cs, false, false);
#pragma unroll
    for (int i = 0; i < 8; ++i) lrow[i] = lrow[i] * fscale[i] + cs[i];

    v16h bv0 = load_b_frag(vb, NS, 0,  s0, lane);
    v16h bv1 = load_b_frag(vb, NS, 16, s0, lane);
    v16h bv2 = load_b_frag(vb, NS, 32, s0, lane);
    v16h bv3 = load_b_frag(vb, NS, 48, s0, lane);
    o0 = __builtin_amdgcn_wmma_f32_16x16x32_f16(false, ap, false, bv0, (short)0, o0, false, false);
    o1 = __builtin_amdgcn_wmma_f32_16x16x32_f16(false, ap, false, bv1, (short)0, o1, false, false);
    o2 = __builtin_amdgcn_wmma_f32_16x16x32_f16(false, ap, false, bv2, (short)0, o2, false, false);
    o3 = __builtin_amdgcn_wmma_f32_16x16x32_f16(false, ap, false, bv3, (short)0, o3, false, false);
  }

  // --- normalize and scatter into [B,T,C] f16 for the output projection ---
  int b = bh >> 4, h = bh & 15;
#pragma unroll
  for (int i = 0; i < 8; ++i) {
    int t = t0 + rowOff + i;
    float inv = 1.0f / lrow[i];
    size_t base = ((size_t)b * NT + t) * NC + (size_t)h * HD;
    atth[base +  0 + col] = (_Float16)(o0[i] * inv);
    atth[base + 16 + col] = (_Float16)(o1[i] * inv);
    atth[base + 32 + col] = (_Float16)(o2[i] * inv);
    atth[base + 48 + col] = (_Float16)(o3[i] * inv);
  }
}

extern "C" void kernel_launch(void* const* d_in, const int* in_sizes, int n_in,
                              void* d_out, int out_size, void* d_ws, size_t ws_size,
                              hipStream_t stream) {
  const float* x  = (const float*)d_in[0];
  const float* pk = (const float*)d_in[1];
  const float* pv = (const float*)d_in[2];
  const float* Wq = (const float*)d_in[3];
  const float* bq = (const float*)d_in[4];
  const float* Wk = (const float*)d_in[5];
  const float* bk = (const float*)d_in[6];
  const float* Wv = (const float*)d_in[7];
  const float* bv = (const float*)d_in[8];
  const float* Wp = (const float*)d_in[9];
  const float* bp = (const float*)d_in[10];

  float* yOut = (float*)d_out;                         // [B,T,C]
  float* kOut = yOut + (size_t)NM * NC;                // [B,H,S,hd]
  float* vOut = kOut + (size_t)NB * NH * NS * HD;      // [B,H,S,hd]

  _Float16* ws   = (_Float16*)d_ws;
  _Float16* xh   = ws;                                  // M*C
  _Float16* WqT  = xh   + (size_t)NM * NC;              // C*C each
  _Float16* WkT  = WqT  + (size_t)NC * NC;
  _Float16* WvT  = WkT  + (size_t)NC * NC;
  _Float16* WpT  = WvT  + (size_t)NC * NC;
  _Float16* qh   = WpT  + (size_t)NC * NC;              // B*H*T*hd
  _Float16* khh  = qh   + (size_t)NB * NH * NT * HD;    // B*H*S*hd
  _Float16* vth  = khh  + (size_t)NB * NH * NS * HD;    // B*H*hd*S
  _Float16* atth = vth  + (size_t)NB * NH * NS * HD;    // M*C

  // stage 0: precision conversion + weight transposes
  convert_x_kernel<<<(NM * NC) / 256, 256, 0, stream>>>(x, xh);
  transpose_w_kernel<<<(NC * NC) / 256, 256, 0, stream>>>(Wq, WqT);
  transpose_w_kernel<<<(NC * NC) / 256, 256, 0, stream>>>(Wk, WkT);
  transpose_w_kernel<<<(NC * NC) / 256, 256, 0, stream>>>(Wv, WvT);
  transpose_w_kernel<<<(NC * NC) / 256, 256, 0, stream>>>(Wp, WpT);
  copy_past_kernel<<<(NB * NH * NP * HD) / 256, 256, 0, stream>>>(pk, pv, kOut,
                                                                  vOut, khh, vth);

  // stage 1: QKV projections (WMMA, 16x64 strip per wave)
  const int gemmBlocks = ((NM / 16) * (NC / 64) * 32) / 256;  // 512
  gemm_kernel<0><<<gemmBlocks, 256, 0, stream>>>(xh, WqT, bq, nullptr, qh);
  gemm_kernel<1><<<gemmBlocks, 256, 0, stream>>>(xh, WkT, bk, kOut, khh);
  gemm_kernel<2><<<gemmBlocks, 256, 0, stream>>>(xh, WvT, bv, vOut, vth);

  // stage 2: flash attention (WMMA + DPP softmax + LDS staging)
  attn_kernel<<<(NB * NH * (NT / 16)) / 8, 256, 0, stream>>>(qh, khh, vth, atth);

  // stage 3: output projection (WMMA)
  gemm_kernel<3><<<gemmBlocks, 256, 0, stream>>>(atth, WpT, bp, yOut, nullptr);
}
